// Block_58394375356873
// MI455X (gfx1250) — matrix operand order repaired and checked by reading, not compiled, not run on hardware
//
#include <hip/hip_runtime.h>
#include <hip/hip_bf16.h>
#include <math.h>

// ---------------- problem constants ----------------
#define B_    64
#define N_    196
#define C_    768
#define H_    12
#define HD_   64
#define TOPK_ 16
#define HID_  3072
#define M_    (B_ * N_)        // 12544 = 98 * 128 (exact tiling, no M edge)
#define SCALE_ 0.125f          // HD^-0.5

typedef unsigned short u16;
typedef __attribute__((ext_vector_type(16))) __bf16 v16bf;
typedef __attribute__((ext_vector_type(8)))  float  v8f;

union Frag { uint4 u[2]; v16bf v; };

// ---- async global->LDS copy (CDNA5 GLOBAL_LOAD_ASYNC_TO_LDS_B128), guarded ----
#if defined(__gfx1250__) && __has_builtin(__builtin_amdgcn_global_load_async_to_lds_b128) && \
    __has_builtin(__builtin_amdgcn_s_wait_asynccnt)
#define HAS_ASYNC_LDS 1
typedef int v4i __attribute__((vector_size(16)));
typedef __attribute__((address_space(1))) v4i* g_v4i_p;   // pointer to AS1 int4
typedef __attribute__((address_space(3))) v4i* l_v4i_p;   // pointer to AS3 int4
#else
#define HAS_ASYNC_LDS 0
#endif

__device__ __forceinline__ void async_ld128(u16* lds, const u16* g) {
#if HAS_ASYNC_LDS
  __builtin_amdgcn_global_load_async_to_lds_b128((g_v4i_p)g, (l_v4i_p)lds, 0, 0);
#else
  *(uint4*)lds = *(const uint4*)g;
#endif
}
__device__ __forceinline__ void async_wait0() {
#if HAS_ASYNC_LDS
  __builtin_amdgcn_s_wait_asynccnt(0);
#endif
}

__device__ __forceinline__ u16 f2bf(float f) {
  unsigned int u = __float_as_uint(f);
  u += 0x7FFFu + ((u >> 16) & 1u);   // round-to-nearest-even
  return (u16)(u >> 16);
}

// A-matrix (16-bit, 16x32) fragment per ISA layout:
// lanes 0-15: M=lane, K = {kbase+0..7, kbase+16..23}
// lanes 16-31: M=lane-16, K = {kbase+8..15, kbase+24..31}
__device__ __forceinline__ v16bf lds_a_frag(const u16* lds, int row_base, int stride,
                                            int kbase, int lane) {
  int m = lane & 15, half = lane >> 4;
  const u16* p = lds + (row_base + m) * stride + kbase + half * 8;
  Frag f;
  f.u[0] = *(const uint4*)p;
  f.u[1] = *(const uint4*)(p + 16);
  return f.v;
}

// B-matrix (16-bit, 32x16) fragment: lane = output column,
// lanes 0-15 hold K={kbase+0..7, kbase+16..23}, lanes 16-31 hold K={kbase+8..15, kbase+24..31}.
__device__ __forceinline__ v16bf lds_b_frag(const u16* lds, int col_base, int stride,
                                            int kbase, int lane) {
  int n = lane & 15, half = lane >> 4;
  const u16* p = lds + (col_base + n) * stride + kbase + half * 8;
  Frag f;
  f.u[0] = *(const uint4*)p;
  f.u[1] = *(const uint4*)(p + 16);
  return f.v;
}

// ---------------- fp32 -> bf16 weight conversion ----------------
__global__ void cvt_kernel(const float* __restrict__ src, u16* __restrict__ dst, int n) {
  int i = blockIdx.x * 256 + threadIdx.x;
  if (i < n) dst[i] = f2bf(src[i]);
}

// ---------------- LayerNorm (768 per row) -> bf16 ----------------
__global__ __launch_bounds__(256)
void ln_kernel(const float* __restrict__ x, const float* __restrict__ g,
               const float* __restrict__ bt, u16* __restrict__ out) {
  __shared__ float r1[256], r2[256];
  __shared__ float stat[2];
  const int row = blockIdx.x, tid = threadIdx.x;
  const float* p = x + (size_t)row * C_;
  float v0 = p[tid], v1 = p[tid + 256], v2 = p[tid + 512];
  r1[tid] = v0 + v1 + v2;
  r2[tid] = v0 * v0 + v1 * v1 + v2 * v2;
  __syncthreads();
  for (int o = 128; o > 0; o >>= 1) {
    if (tid < o) { r1[tid] += r1[tid + o]; r2[tid] += r2[tid + o]; }
    __syncthreads();
  }
  if (tid == 0) {
    float mu  = r1[0] * (1.0f / C_);
    float var = r2[0] * (1.0f / C_) - mu * mu;
    stat[0] = mu;
    stat[1] = rsqrtf(var + 1e-5f);
  }
  __syncthreads();
  float mu = stat[0], rs = stat[1];
  u16* q = out + (size_t)row * C_;
  q[tid]       = f2bf((v0 - mu) * rs * g[tid]       + bt[tid]);
  q[tid + 256] = f2bf((v1 - mu) * rs * g[tid + 256] + bt[tid + 256]);
  q[tid + 512] = f2bf((v2 - mu) * rs * g[tid + 512] + bt[tid + 512]);
}

// ---------------- tiled WMMA GEMM: Out = A[M,K] * W[N,K]^T ----------------
// Double-buffered LDS, async global->LDS fill overlapped with WMMA compute.
// MODE 0: qk -> split into scaled q / k, scattered [B,H,N,HD] bf16
// MODE 1: proj -> fp32 out = resid + bias + acc
// MODE 2: fc1  -> bf16 out = gelu(acc + bias)
// MODE 3: fc2  -> fp32 out = resid + bias + acc
template <int MODE>
__global__ __launch_bounds__(256)
void gemm_wmma(const u16* __restrict__ A, const u16* __restrict__ W,
               int K, int Nout,
               const float* __restrict__ bias, const float* __restrict__ resid,
               float* __restrict__ outF, u16* __restrict__ outH,
               u16* __restrict__ outQ, u16* __restrict__ outK) {
  __shared__ __align__(16) u16 As[2][128 * 32];
  __shared__ __align__(16) u16 Bs[2][128 * 32];
  const int tid = threadIdx.x;
  const int lane = tid & 31, wave = tid >> 5;
  const int wm = wave >> 1, wn = wave & 1;          // 4x2 wave grid
  const int m0 = blockIdx.x * 128, n0 = blockIdx.y * 128;

  const int row = tid >> 1, seg = (tid & 1) * 16;   // tile-fill assignment
  const u16* gA = A + (size_t)(m0 + row) * K + seg; // + kt*32
  const u16* gB = W + (size_t)(n0 + row) * K + seg;

  v8f acc[2][4] = {};
  const int nkt = K >> 5;                           // BK = 32

  // prologue: start fill of tile 0 into buffer 0
  {
    async_ld128(&As[0][row * 32 + seg],     gA);
    async_ld128(&As[0][row * 32 + seg + 8], gA + 8);
    async_ld128(&Bs[0][row * 32 + seg],     gB);
    async_ld128(&Bs[0][row * 32 + seg + 8], gB + 8);
  }

  for (int kt = 0; kt < nkt; ++kt) {
    const int cur = kt & 1, nxt = cur ^ 1;
    async_wait0();          // my 4 async fills of buffer `cur` have landed in LDS
    __syncthreads();        // everyone's fills landed; everyone done reading `nxt`
    if (kt + 1 < nkt) {     // overlap: fill next tile while computing current
      const u16* pa = gA + (size_t)(kt + 1) * 32;
      const u16* pb = gB + (size_t)(kt + 1) * 32;
      async_ld128(&As[nxt][row * 32 + seg],     pa);
      async_ld128(&As[nxt][row * 32 + seg + 8], pa + 8);
      async_ld128(&Bs[nxt][row * 32 + seg],     pb);
      async_ld128(&Bs[nxt][row * 32 + seg + 8], pb + 8);
      if (kt + 2 < nkt) __builtin_prefetch(pa + 32, 0, 0);
    }
    v16bf a0 = lds_a_frag(As[cur], wm * 32,      32, 0, lane);
    v16bf a1 = lds_a_frag(As[cur], wm * 32 + 16, 32, 0, lane);
#pragma unroll
    for (int nf = 0; nf < 4; ++nf) {
      v16bf bb = lds_b_frag(Bs[cur], wn * 64 + nf * 16, 32, 0, lane);
      acc[0][nf] = __builtin_amdgcn_wmma_f32_16x16x32_bf16(
          false, a0, false, bb, (short)0, acc[0][nf], false, false);
      acc[1][nf] = __builtin_amdgcn_wmma_f32_16x16x32_bf16(
          false, a1, false, bb, (short)0, acc[1][nf], false, false);
    }
  }
  async_wait0();  // drain — no-op when nothing outstanding

  // epilogue — D layout: vgpr i, lanes 0-15 -> (M=i, N=lane); lanes 16-31 -> (M=i+8)
#pragma unroll
  for (int mf = 0; mf < 2; ++mf) {
#pragma unroll
    for (int nf = 0; nf < 4; ++nf) {
#pragma unroll
      for (int i = 0; i < 8; ++i) {
        int r = m0 + wm * 32 + mf * 16 + i + (lane >> 4) * 8;
        int c = n0 + wn * 64 + nf * 16 + (lane & 15);
        float v = acc[mf][nf][i];
        if (MODE == 0) {
          int b = r / N_, n = r % N_;
          if (c < C_) {
            int h = c >> 6, hd = c & 63;
            outQ[(((size_t)(b * H_ + h)) * N_ + n) * HD_ + hd] = f2bf(v * SCALE_);
          } else {
            int c2 = c - C_;
            int h = c2 >> 6, hd = c2 & 63;
            outK[(((size_t)(b * H_ + h)) * N_ + n) * HD_ + hd] = f2bf(v);
          }
        } else if (MODE == 1 || MODE == 3) {
          size_t idx = (size_t)r * Nout + c;
          outF[idx] = resid[idx] + v + bias[c];
        } else {  // MODE 2: exact GELU -> bf16
          float t = v + bias[c];
          float gl = 0.5f * t * (1.0f + erff(t * 0.70710678118654752f));
          outH[(size_t)r * Nout + c] = f2bf(gl);
        }
      }
    }
  }
}

// ---------------- fused scores + top-16 + softmax ----------------
// One workgroup = one (b, h, 16-row tile). Scores stay in LDS; no [B,H,N,N] buffer.
__global__ __launch_bounds__(256)
void attn_topk(const u16* __restrict__ qb, const u16* __restrict__ kb,
               u16* __restrict__ ab) {
  __shared__ __align__(16) u16 Qs[16 * 64];        // 2 KB
  __shared__ __align__(16) u16 Ks[8][16 * 64];     // 16 KB (wave-private tiles)
  __shared__ float S[16][208];                     // 13 KB score strip

  const int tid = threadIdx.x, lane = tid & 31, wave = tid >> 5;
  const int ntile = blockIdx.x, h = blockIdx.y, b = blockIdx.z;
  const size_t base = ((size_t)(b * H_ + h)) * N_ * HD_;

  if (tid < 128) {                                  // stage Q tile (16x64 bf16)
    int rw = tid >> 3, sg = (tid & 7) * 8;
    int n = ntile * 16 + rw; if (n > N_ - 1) n = N_ - 1;
    *(uint4*)(Qs + rw * 64 + sg) = *(const uint4*)(qb + base + (size_t)n * HD_ + sg);
  }
  __syncthreads();

  for (int tt = wave; tt < 13; tt += 8) {           // 13 K-tiles cover 208 cols
#pragma unroll
    for (int j = 0; j < 4; ++j) {                   // async fill wave-private K tile
      int id = lane + 32 * j;
      int rw = id >> 3, sg = (id & 7) * 8;
      int m = tt * 16 + rw; if (m > N_ - 1) m = N_ - 1;
      async_ld128(&Ks[wave][rw * 64 + sg], kb + base + (size_t)m * HD_ + sg);
    }
    async_wait0();                                  // wave-private: no barrier needed
    v8f acc = {};
#pragma unroll
    for (int ks = 0; ks < 2; ++ks) {                // HD=64 -> two K=32 WMMAs
      v16bf aq = lds_a_frag(Qs, 0, 64, ks * 32, lane);
      v16bf bk = lds_b_frag(&Ks[wave][0], 0, 64, ks * 32, lane);
      acc = __builtin_amdgcn_wmma_f32_16x16x32_bf16(
          false, aq, false, bk, (short)0, acc, false, false);
    }
#pragma unroll
    for (int i = 0; i < 8; ++i) {
      int rr = i + (lane >> 4) * 8, cc = lane & 15;
      S[rr][tt * 16 + cc] = acc[i];
    }
  }
  __syncthreads();

  // top-16 (descending, lowest-index tie-break like lax.top_k) + softmax.
  for (int rsel = 0; rsel < 2; ++rsel) {
    int rr = wave * 2 + rsel;
    int gn = ntile * 16 + rr;
    if (gn >= N_) continue;
    float m0 = 0.f, sum = 0.f, myval = 0.f;
    for (int sel = 0; sel < TOPK_; ++sel) {
      float best = -1e30f; int bidx = 0x7fffffff;
#pragma unroll
      for (int j = 0; j < 7; ++j) {
        int col = lane + 32 * j;
        if (col < N_) {
          float v = S[rr][col];
          if (v > best) { best = v; bidx = col; }
        }
      }
#pragma unroll
      for (int o = 16; o > 0; o >>= 1) {
        float ov = __shfl_xor(best, o, 32);
        int   oi = __shfl_xor(bidx, o, 32);
        if (ov > best || (ov == best && oi < bidx)) { best = ov; bidx = oi; }
      }
      if (sel == 0) m0 = best;
      sum += expf(best - m0);
      if (lane == sel) myval = best;
      if (lane == 0) S[rr][bidx] = -1e30f;          // knock out, in-order wave LDS
    }
    if (lane < TOPK_) {
      float a = expf(myval - m0) / sum;
      ab[((size_t)(b * N_ + gn)) * (H_ * TOPK_) + h * TOPK_ + lane] = f2bf(a);
    }
  }
}

// ---------------- driver ----------------
extern "C" void kernel_launch(void* const* d_in, const int* in_sizes, int n_in,
                              void* d_out, int out_size, void* d_ws, size_t ws_size,
                              hipStream_t stream) {
  const float* x      = (const float*)d_in[0];
  const float* n1g    = (const float*)d_in[1];
  const float* n1b    = (const float*)d_in[2];
  const float* qk_w   = (const float*)d_in[3];
  const float* proj_w = (const float*)d_in[4];
  const float* proj_b = (const float*)d_in[5];
  const float* n2g    = (const float*)d_in[6];
  const float* n2b    = (const float*)d_in[7];
  const float* fc1_w  = (const float*)d_in[8];
  const float* fc1_b  = (const float*)d_in[9];
  const float* fc2_w  = (const float*)d_in[10];
  const float* fc2_b  = (const float*)d_in[11];
  float* out = (float*)d_out;
  (void)in_sizes; (void)n_in; (void)out_size; (void)ws_size;

  char* ws = (char*)d_ws;
  size_t off = 0;
  auto alloc = [&](size_t bytes) -> char* {
    char* p = ws + off;
    off += (bytes + 255) & ~(size_t)255;
    return p;
  };
  u16*   hb  = (u16*)  alloc((size_t)M_ * C_ * 2);            // ln1(x) bf16
  u16*   qb  = (u16*)  alloc((size_t)B_ * H_ * N_ * HD_ * 2); // q bf16 (scaled)
  u16*   kb  = (u16*)  alloc((size_t)B_ * H_ * N_ * HD_ * 2); // k bf16
  u16*   ab  = (u16*)  alloc((size_t)M_ * H_ * TOPK_ * 2);    // softmax(topk) bf16
  float* x1  = (float*)alloc((size_t)M_ * C_ * 4);            // x + attn branch
  u16*   h2b = (u16*)  alloc((size_t)M_ * C_ * 2);            // ln2(x1) bf16
  u16*   m1b = (u16*)  alloc((size_t)M_ * HID_ * 2);          // gelu(fc1) bf16
  u16*   wqk = (u16*)  alloc((size_t)2 * C_ * C_ * 2);
  u16*   wpj = (u16*)  alloc((size_t)C_ * H_ * TOPK_ * 2);
  u16*   wf1 = (u16*)  alloc((size_t)HID_ * C_ * 2);
  u16*   wf2 = (u16*)  alloc((size_t)C_ * HID_ * 2);

  int n;
  n = 2 * C_ * C_;     cvt_kernel<<<(n + 255) / 256, 256, 0, stream>>>(qk_w,  wqk, n);
  n = C_ * H_ * TOPK_; cvt_kernel<<<(n + 255) / 256, 256, 0, stream>>>(proj_w, wpj, n);
  n = HID_ * C_;       cvt_kernel<<<(n + 255) / 256, 256, 0, stream>>>(fc1_w, wf1, n);
  n = C_ * HID_;       cvt_kernel<<<(n + 255) / 256, 256, 0, stream>>>(fc2_w, wf2, n);

  ln_kernel<<<M_, 256, 0, stream>>>(x, n1g, n1b, hb);

  gemm_wmma<0><<<dim3(M_ / 128, (2 * C_) / 128), 256, 0, stream>>>(
      hb, wqk, C_, 2 * C_, nullptr, nullptr, nullptr, nullptr, qb, kb);

  attn_topk<<<dim3(13, H_, B_), 256, 0, stream>>>(qb, kb, ab);

  gemm_wmma<1><<<dim3(M_ / 128, C_ / 128), 256, 0, stream>>>(
      ab, wpj, H_ * TOPK_, C_, proj_b, x, x1, nullptr, nullptr, nullptr);

  ln_kernel<<<M_, 256, 0, stream>>>(x1, n2g, n2b, h2b);

  gemm_wmma<2><<<dim3(M_ / 128, HID_ / 128), 256, 0, stream>>>(
      h2b, wf1, C_, HID_, fc1_b, nullptr, nullptr, m1b, nullptr, nullptr);

  gemm_wmma<3><<<dim3(M_ / 128, C_ / 128), 256, 0, stream>>>(
      m1b, wf2, HID_, C_, fc2_b, x1, out, nullptr, nullptr, nullptr);
}
